// HierarchicalGraphNet_15796889715336
// MI455X (gfx1250) — matrix operand order repaired and checked by reading, not compiled, not run on hardware
//
#include <hip/hip_runtime.h>
#include <stdint.h>

// Hierarchical Graph U-Net (EdgePooling) for MI455X / gfx1250.
// GEMMs: V_WMMA_F32_16X16X32_F16 (f16 in, f32 accum). Weights pre-converted
// once per launch to column-major f16 (B loads = 2x global b128/k-step);
// X row-panel staged in LDS as f16, shared by all 8 waves in a block
// (A loads = 2x ds b128/k-step). Segment ops use global f32 atomics
// (working set is L2-resident on 192MB L2); JAX-stable ordering reproduced
// via composite u64 keys + bitonic sort; greedy matching is a single-thread
// scan (inherently sequential in the reference).

#define NN 20000
#define EE 200000
#define CC 128
#define SORT_N 262144          // next pow2 >= EE
#define NEGC (-1e30f)
#define SENT_KEY 400000000u    // NN*NN sentinel for edge-dedup keys

typedef __attribute__((ext_vector_type(16))) _Float16 v16h;
typedef __attribute__((ext_vector_type(8)))  _Float16 v8h;
typedef __attribute__((ext_vector_type(8)))  float    v8f;

__device__ __forceinline__ unsigned ordU(float f) {
  unsigned u = __float_as_uint(f);
  return (u & 0x80000000u) ? ~u : (u | 0x80000000u);
}
__device__ __forceinline__ float invOrdU(unsigned o) {
  return __uint_as_float((o & 0x80000000u) ? (o ^ 0x80000000u) : ~o);
}

// ---- weight pre-pass: W (128x128 f32 row-major) -> Wt (col-major f16) ------
__global__ void k_cvt_w(const float* __restrict__ W, _Float16* __restrict__ Wt) {
  int t = blockIdx.x * blockDim.x + threadIdx.x;
  if (t >= CC * CC) return;
  int n = t >> 7, k = t & 127;
  Wt[t] = (_Float16)W[(size_t)k * CC + n];   // Wt[n*128+k] = W[k][n]
}

// ---------------- WMMA GEMM: Y[r,c] = sum_k X[r,k]*Wt[c,k] (+bias)(+relu) ---
// X: nrows x 128 f32 row-major; Wt: 128x128 f16 column-major (Wt[n][k]).
// One wave = one 16x16 tile; block = 256 thr = 8 waves = 16 rows x 128 cols.
// A panel (16x128) staged once in LDS as f16 and shared by all 8 waves.
__global__ void k_gemm128(const float* __restrict__ X, const _Float16* __restrict__ Wt,
                          const float* __restrict__ bias, float* __restrict__ Y,
                          int relu) {
  __shared__ _Float16 As[16 * CC];           // 4 KB
  const int lane = threadIdx.x & 31;
  const int wave = threadIdx.x >> 5;
  const int row0 = blockIdx.x << 4;
  const int col0 = wave << 4;
  const int m    = lane & 15;
  const int hlf  = lane >> 4;

  // cooperative stage: each thread converts 8 consecutive floats
  {
    int off = threadIdx.x * 8;               // 0..2040
    const float4* src = (const float4*)(X + (size_t)row0 * CC + off);
    float4 p0 = src[0], p1 = src[1];
    _Float16* d = As + off;
    d[0] = (_Float16)p0.x; d[1] = (_Float16)p0.y;
    d[2] = (_Float16)p0.z; d[3] = (_Float16)p0.w;
    d[4] = (_Float16)p1.x; d[5] = (_Float16)p1.y;
    d[6] = (_Float16)p1.z; d[7] = (_Float16)p1.w;
  }
  __syncthreads();

  v8f acc = {};
#pragma unroll
  for (int k0 = 0; k0 < CC; k0 += 32) {
    // A fragment (16x32 f16 layout, ISA 7.12.2): lane (m,hlf) needs
    // X[m][k0+hlf*8 .. +7] and X[m][k0+16+hlf*8 .. +7]  -> 2x ds b128
    const _Float16* ap = As + m * CC + k0 + (hlf << 3);
    v8h a0 = *(const v8h*)(ap);
    v8h a1 = *(const v8h*)(ap + 16);
    v16h a = __builtin_shufflevector(a0, a1, 0, 1, 2, 3, 4, 5, 6, 7,
                                     8, 9, 10, 11, 12, 13, 14, 15);
    // B fragment (32x16 f16): lane col = col0+m, same K order as A
    // -> Wt[col][k0+hlf*8 .. +7] and Wt[col][k0+16+hlf*8 .. +7]  -> 2x global b128
    const _Float16* bp = Wt + (size_t)(col0 + m) * CC + k0 + (hlf << 3);
    v8h b0 = *(const v8h*)(bp);
    v8h b1 = *(const v8h*)(bp + 16);
    v16h b = __builtin_shufflevector(b0, b1, 0, 1, 2, 3, 4, 5, 6, 7,
                                     8, 9, 10, 11, 12, 13, 14, 15);
    acc = __builtin_amdgcn_wmma_f32_16x16x32_f16(false, a, false, b,
                                                 (short)0, acc, false, false);
  }
  // D layout: VGPR r -> row r + 8*hlf, col = m.
#pragma unroll
  for (int r = 0; r < 8; ++r) {
    int row = row0 + r + (hlf << 3);
    int col = col0 + m;
    float v = acc[r];
    if (bias) v += bias[col];
    if (relu) v = fmaxf(v, 0.0f);
    Y[(size_t)row * CC + col] = v;
  }
}

// ---------------- elementwise / fill ----------------------------------------
__global__ void k_fill_f32(float* p, float v, int n) {
  int i = blockIdx.x * blockDim.x + threadIdx.x;
  if (i < n) p[i] = v;
}
__global__ void k_fill_i32(int* p, int v, int n) {
  int i = blockIdx.x * blockDim.x + threadIdx.x;
  if (i < n) p[i] = v;
}
__global__ void k_fill_u32(unsigned* p, unsigned v, int n) {
  int i = blockIdx.x * blockDim.x + threadIdx.x;
  if (i < n) p[i] = v;
}
__global__ void k_iota_i32(int* p, int n) {
  int i = blockIdx.x * blockDim.x + threadIdx.x;
  if (i < n) p[i] = i;
}

// ---------------- GCN pieces ------------------------------------------------
__global__ void k_deg(const int* __restrict__ dst, const int* __restrict__ ev,
                      float* deg, int ne) {
  int e = blockIdx.x * blockDim.x + threadIdx.x;
  if (e < ne && ev[e]) atomicAdd(&deg[dst[e]], 1.0f);
}
__global__ void k_rsqrt(float* d, int n) {
  int i = blockIdx.x * blockDim.x + threadIdx.x;
  if (i < n) d[i] = rsqrtf(d[i]);
}
__global__ void k_agg(const float* __restrict__ h, const int* __restrict__ src,
                      const int* __restrict__ dst, const int* __restrict__ ev,
                      const float* __restrict__ dinv, float* agg, int ne) {
  int t = blockIdx.x * blockDim.x + threadIdx.x;
  if (t >= ne * CC) return;
  int e = t >> 7, c = t & 127;
  if (!ev[e]) return;
  int s = src[e], d = dst[e];
  atomicAdd(&agg[(size_t)d * CC + c], h[(size_t)s * CC + c] * dinv[s] * dinv[d]);
}
__global__ void k_finish(const float* __restrict__ h, const float* __restrict__ agg,
                         const float* __restrict__ dinv, const float* __restrict__ bias,
                         float* out, int relu) {
  int t = blockIdx.x * blockDim.x + threadIdx.x;
  if (t >= NN * CC) return;
  int i = t >> 7, c = t & 127;
  float v = agg[t] + h[t] * dinv[i] * dinv[i] + bias[c];
  if (relu) v = fmaxf(v, 0.0f);
  out[t] = v;
}

// ---------------- EdgePooling pieces ----------------------------------------
__global__ void k_escore(const float* __restrict__ x, const int* __restrict__ src,
                         const int* __restrict__ dst, const int* __restrict__ ev,
                         const float* __restrict__ wp, const float* __restrict__ bp,
                         float* esc, int ne) {
  int e = blockIdx.x * blockDim.x + threadIdx.x;
  if (e >= ne) return;
  if (!ev[e]) { esc[e] = NEGC; return; }
  const float* xs = x + (size_t)src[e] * CC;
  const float* xd = x + (size_t)dst[e] * CC;
  float acc = 0.f;
#pragma unroll 4
  for (int c = 0; c < CC; ++c) acc += xs[c] * wp[c] + xd[c] * wp[CC + c];
  esc[e] = acc + bp[0];
}
__global__ void k_segmax(const float* __restrict__ esc, const int* __restrict__ dst,
                         const int* __restrict__ ev, unsigned* mOrd, int ne) {
  int e = blockIdx.x * blockDim.x + threadIdx.x;
  if (e < ne && ev[e]) atomicMax(&mOrd[dst[e]], ordU(esc[e]));
}
__global__ void k_expsum(const float* __restrict__ esc, const int* __restrict__ dst,
                         const int* __restrict__ ev, const unsigned* __restrict__ mOrd,
                         float* ex, float* ssum, int ne) {
  int e = blockIdx.x * blockDim.x + threadIdx.x;
  if (e >= ne) return;
  if (!ev[e]) { ex[e] = 0.f; return; }
  int d = dst[e];
  float v = __expf(esc[e] - invOrdU(mOrd[d]));
  ex[e] = v;
  atomicAdd(&ssum[d], v);
}
__global__ void k_score(float* ex, const float* __restrict__ ssum,
                        const int* __restrict__ dst, const int* __restrict__ ev, int ne) {
  int e = blockIdx.x * blockDim.x + threadIdx.x;
  if (e >= ne || !ev[e]) return;
  float s = ssum[dst[e]];
  s = (s > 0.f) ? s : 1.f;
  ex[e] = ex[e] / s + 0.5f;
}
// composite key: (~ordered(score)) << 32 | idx  -> ascending sort == stable
// descending-score argsort.  Padding / invalid sort last (top bit set).
__global__ void k_key1(const float* __restrict__ score, const int* __restrict__ ev,
                       unsigned long long* key, int ne) {
  int t = blockIdx.x * blockDim.x + threadIdx.x;
  if (t >= SORT_N) return;
  if (t < ne) {
    float f = ev[t] ? score[t] : -__builtin_inff();
    key[t] = ((unsigned long long)(~ordU(f)) << 32) | (unsigned)t;
  } else {
    key[t] = ~0ull;
  }
}
__global__ void k_bitonic(unsigned long long* key, int j, int k) {
  int i = blockIdx.x * blockDim.x + threadIdx.x;
  int l = i ^ j;
  if (l > i) {
    unsigned long long a = key[i], b = key[l];
    bool asc = ((i & k) == 0);
    if ((a > b) == asc) { key[i] = b; key[l] = a; }
  }
}
// Sequential greedy matching in sorted order (exact reference semantics).
__global__ void k_greedy(const unsigned long long* __restrict__ key,
                         const int* __restrict__ src, const int* __restrict__ dst,
                         const float* __restrict__ score,
                         int* avail, int* cluster, float* sscore) {
  if (threadIdx.x != 0 || blockIdx.x != 0) return;
  for (int t = 0; t < SORT_N; ++t) {
    unsigned long long kk = key[t];
    unsigned hi = (unsigned)(kk >> 32);
    if (hi & 0x80000000u) break;         // first invalid (-inf) / padding key
    unsigned idx = (unsigned)kk;
    int si = src[idx], di = dst[idx];
    if (avail[si] && avail[di]) {
      avail[si] = 0; avail[di] = 0;
      cluster[di] = si;
      sscore[si] = score[idx];
    }
  }
}
__global__ void k_poolsum(const float* __restrict__ x, const int* __restrict__ cluster,
                          float* newx) {
  int t = blockIdx.x * blockDim.x + threadIdx.x;
  if (t >= NN * CC) return;
  int i = t >> 7, c = t & 127;
  atomicAdd(&newx[(size_t)cluster[i] * CC + c], x[t]);
}
__global__ void k_poolscale(const float* __restrict__ newx,
                            const float* __restrict__ sscore, float* xout) {
  int t = blockIdx.x * blockDim.x + threadIdx.x;
  if (t >= NN * CC) return;
  xout[t] = newx[t] * sscore[t >> 7];
}
__global__ void k_remap(const int* __restrict__ src, const int* __restrict__ dst,
                        const int* __restrict__ ev, const int* __restrict__ cluster,
                        int* nsrc, int* ndst, unsigned long long* key2, int ne) {
  int t = blockIdx.x * blockDim.x + threadIdx.x;
  if (t >= SORT_N) return;
  if (t < ne) {
    int rs = cluster[src[t]], rd = cluster[dst[t]];
    nsrc[t] = rs; ndst[t] = rd;
    bool nontriv = ev[t] && (rs != rd);
    unsigned hi = nontriv ? (unsigned)(rs * NN + rd) : SENT_KEY;
    key2[t] = ((unsigned long long)hi << 32) | (unsigned)t;
  } else {
    key2[t] = ~0ull;
  }
}
__global__ void k_dedup(const unsigned long long* __restrict__ key2, int* nev, int ne) {
  int t = blockIdx.x * blockDim.x + threadIdx.x;
  if (t >= SORT_N) return;
  unsigned long long kk = key2[t];
  unsigned idx = (unsigned)kk;
  if (idx >= (unsigned)ne) return;       // padding
  unsigned hi = (unsigned)(kk >> 32);
  if (hi >= SENT_KEY) { nev[idx] = 0; return; }
  unsigned ph = (t == 0) ? 0xFFFFFFFFu : (unsigned)(key2[t - 1] >> 32);
  nev[idx] = (hi != ph) ? 1 : 0;
}
__global__ void k_unpool(const float* __restrict__ xs, const float* __restrict__ xup,
                         const int* __restrict__ cluster, const float* __restrict__ sscore,
                         float* out) {
  int t = blockIdx.x * blockDim.x + threadIdx.x;
  if (t >= NN * CC) return;
  int i = t >> 7, c = t & 127;
  int j = cluster[i];
  out[t] = xs[t] + xup[(size_t)j * CC + c] / sscore[j];
}

// ---------------- host orchestration ----------------------------------------
extern "C" void kernel_launch(void* const* d_in, const int* in_sizes, int n_in,
                              void* d_out, int out_size, void* d_ws, size_t ws_size,
                              hipStream_t stream) {
  (void)in_sizes; (void)n_in; (void)out_size; (void)ws_size;
  const float* x_in   = (const float*)d_in[0];
  const int*   src_in = (const int*)d_in[1];
  const int*   dst_in = (const int*)d_in[2];
  const float* w_pre  = (const float*)d_in[3];
  const float* b_pre  = (const float*)d_in[4];
  const float* w_up[4]   = {(const float*)d_in[5], (const float*)d_in[7],
                            (const float*)d_in[9], (const float*)d_in[11]};
  const float* b_up[4]   = {(const float*)d_in[6], (const float*)d_in[8],
                            (const float*)d_in[10], (const float*)d_in[12]};
  const float* w_pool[3] = {(const float*)d_in[13], (const float*)d_in[17],
                            (const float*)d_in[21]};
  const float* b_pool[3] = {(const float*)d_in[14], (const float*)d_in[18],
                            (const float*)d_in[22]};
  const float* w_down[3] = {(const float*)d_in[15], (const float*)d_in[19],
                            (const float*)d_in[23]};
  const float* b_down[3] = {(const float*)d_in[16], (const float*)d_in[20],
                            (const float*)d_in[24]};

  // ---- carve workspace (~86 MB) ----
  char* cur = (char*)d_ws;
  auto carve = [&](size_t bytes) -> void* {
    void* p = (void*)cur;
    cur += (bytes + 255) & ~(size_t)255;
    return p;
  };
  const size_t NCb = (size_t)NN * CC * sizeof(float);
  float* x_cur = (float*)carve(NCb);
  float* xtmp  = (float*)carve(NCb);
  float* hbuf  = (float*)carve(NCb);
  float* agg   = (float*)carve(NCb);
  float* xs[3]; for (int i = 0; i < 3; ++i) xs[i] = (float*)carve(NCb);
  float*    deg  = (float*)carve(NN * 4);
  unsigned* mOrd = (unsigned*)carve(NN * 4);
  float*    ssum = (float*)carve(NN * 4);
  int*      avail= (int*)carve(NN * 4);
  float* sscr[3]; int* clus[3];
  for (int i = 0; i < 3; ++i) { sscr[i] = (float*)carve(NN * 4); clus[i] = (int*)carve(NN * 4); }
  float* esc   = (float*)carve(EE * 4);
  float* score = (float*)carve(EE * 4);
  unsigned long long* keyb = (unsigned long long*)carve((size_t)SORT_N * 8);
  int* esrc[4]; int* edst[4]; int* eev[4];
  for (int i = 0; i < 4; ++i) {
    esrc[i] = (int*)carve(EE * 4); edst[i] = (int*)carve(EE * 4); eev[i] = (int*)carve(EE * 4);
  }
  // f16 column-major weight copies (32 KB each)
  const size_t Wb = (size_t)CC * CC * sizeof(_Float16);
  _Float16* wt_pre = (_Float16*)carve(Wb);
  _Float16* wt_up[4]; for (int i = 0; i < 4; ++i) wt_up[i] = (_Float16*)carve(Wb);
  _Float16* wt_down[3]; for (int i = 0; i < 3; ++i) wt_down[i] = (_Float16*)carve(Wb);

  const int B   = 256;
  const int gNC = (NN * CC) / B;          // 10000
  const int gEC = (EE * CC) / B;          // 100000
  const int gE  = (EE + B - 1) / B;
  const int gN  = (NN + B - 1) / B;
  const int gS  = SORT_N / B;
  const int gG  = NN / 16;                // 1250 GEMM row-tiles
  const int gW  = (CC * CC) / B;          // 64

  // ---- weight pre-pass: f32 row-major -> f16 col-major ----
  k_cvt_w<<<gW, B, 0, stream>>>(w_pre, wt_pre);
  for (int i = 0; i < 4; ++i) k_cvt_w<<<gW, B, 0, stream>>>(w_up[i], wt_up[i]);
  for (int i = 0; i < 3; ++i) k_cvt_w<<<gW, B, 0, stream>>>(w_down[i], wt_down[i]);

  auto sortU64 = [&](unsigned long long* kb) {
    for (int k = 2; k <= SORT_N; k <<= 1)
      for (int j = k >> 1; j > 0; j >>= 1)
        k_bitonic<<<gS, B, 0, stream>>>(kb, j, k);
  };
  auto gcn = [&](const float* xin, const _Float16* Wt, const float* bias,
                 const int* s, const int* d, const int* ev, float* out, int relu) {
    k_gemm128<<<gG, B, 0, stream>>>(xin, Wt, nullptr, hbuf, 0);
    k_fill_f32<<<gNC, B, 0, stream>>>(agg, 0.f, NN * CC);
    k_fill_f32<<<gN, B, 0, stream>>>(deg, 1.f, NN);
    k_deg<<<gE, B, 0, stream>>>(d, ev, deg, EE);
    k_rsqrt<<<gN, B, 0, stream>>>(deg, NN);
    k_agg<<<gEC, B, 0, stream>>>(hbuf, s, d, ev, deg, agg, EE);
    k_finish<<<gNC, B, 0, stream>>>(hbuf, agg, deg, bias, out, relu);
  };
  auto pool = [&](int lvl) {  // lvl in 1..3 consumes level lvl-1, writes level lvl
    int p = lvl - 1;
    k_escore<<<gE, B, 0, stream>>>(x_cur, esrc[p], edst[p], eev[p], w_pool[p], b_pool[p], esc, EE);
    k_fill_u32<<<gN, B, 0, stream>>>(mOrd, 0u, NN);
    k_segmax<<<gE, B, 0, stream>>>(esc, edst[p], eev[p], mOrd, EE);
    k_fill_f32<<<gN, B, 0, stream>>>(ssum, 0.f, NN);
    k_expsum<<<gE, B, 0, stream>>>(esc, edst[p], eev[p], mOrd, score, ssum, EE);
    k_score<<<gE, B, 0, stream>>>(score, ssum, edst[p], eev[p], EE);
    k_key1<<<gS, B, 0, stream>>>(score, eev[p], keyb, EE);
    sortU64(keyb);
    k_fill_i32<<<gN, B, 0, stream>>>(avail, 1, NN);
    k_iota_i32<<<gN, B, 0, stream>>>(clus[p], NN);
    k_fill_f32<<<gN, B, 0, stream>>>(sscr[p], 1.f, NN);
    k_greedy<<<1, 1, 0, stream>>>(keyb, esrc[p], edst[p], score, avail, clus[p], sscr[p]);
    k_fill_f32<<<gNC, B, 0, stream>>>(agg, 0.f, NN * CC);
    k_poolsum<<<gNC, B, 0, stream>>>(x_cur, clus[p], agg);
    k_poolscale<<<gNC, B, 0, stream>>>(agg, sscr[p], x_cur);
    k_remap<<<gS, B, 0, stream>>>(esrc[p], edst[p], eev[p], clus[p],
                                  esrc[lvl], edst[lvl], keyb, EE);
    sortU64(keyb);
    k_dedup<<<gS, B, 0, stream>>>(keyb, eev[lvl], EE);
  };

  // ---- level 0 edges ----
  hipMemcpyAsync(esrc[0], src_in, EE * 4, hipMemcpyDeviceToDevice, stream);
  hipMemcpyAsync(edst[0], dst_in, EE * 4, hipMemcpyDeviceToDevice, stream);
  k_fill_i32<<<gE, B, 0, stream>>>(eev[0], 1, EE);

  // ---- pre-embed + up path ----
  k_gemm128<<<gG, B, 0, stream>>>(x_in, wt_pre, b_pre, x_cur, 0);
  gcn(x_cur, wt_up[0], b_up[0], esrc[0], edst[0], eev[0], x_cur, 1);
  hipMemcpyAsync(xs[0], x_cur, NCb, hipMemcpyDeviceToDevice, stream);
  for (int lvl = 1; lvl <= 3; ++lvl) {
    pool(lvl);
    gcn(x_cur, wt_up[lvl], b_up[lvl], esrc[lvl], edst[lvl], eev[lvl], x_cur, 1);
    if (lvl < 3) hipMemcpyAsync(xs[lvl], x_cur, NCb, hipMemcpyDeviceToDevice, stream);
  }

  // ---- down path ----
  for (int lvl = 2; lvl >= 0; --lvl) {
    k_unpool<<<gNC, B, 0, stream>>>(xs[lvl], x_cur, clus[lvl], sscr[lvl], xtmp);
    float* out = (lvl > 0) ? x_cur : (float*)d_out;
    gcn(xtmp, wt_down[lvl], b_down[lvl], esrc[lvl], edst[lvl], eev[lvl], out, lvl > 0);
  }
}
